// Sequence_12163347383197
// MI455X (gfx1250) — matrix operand order repaired and checked by reading, not compiled, not run on hardware
//
#include <hip/hip_runtime.h>

// 2-layer LSTM (H=51) persistent recurrent kernel for gfx1250 (MI455X).
// One 128-thread block owns one 16-row batch tile for all T steps; wave g
// (of 4) owns gate-column group {g, g+4, g+8, g+12} of the N=256 padded gate
// matrix, cutting the per-step critical path to 24 WMMAs/wave. h state is
// double-buffered in LDS (1 barrier/step); c state lives in registers in WMMA
// C-fragment layout. All GEMMs use v_wmma_f32_16x16x32_f16.

#define HH 51
#define BLK 128  // 4 waves

typedef _Float16 h8v  __attribute__((ext_vector_type(8)));
typedef _Float16 h16v __attribute__((ext_vector_type(16)));
typedef float    f8v  __attribute__((ext_vector_type(8)));

union H16 { h16v v; h8v h[2]; };
union F8  { f8v  v; float f[8]; };

__device__ __forceinline__ float fsig(float x) {
  float e = __builtin_amdgcn_exp2f(x * -1.4426950408889634f);
  return __builtin_amdgcn_rcpf(1.0f + e);
}
__device__ __forceinline__ float ftanh(float x) {
  return 2.0f * fsig(2.0f * x) - 1.0f;
}

// A-fragment (16x32 f16, MxK): lane holds row m=lane&15; lanes 0-15 carry
// K={0..7,16..23}, lanes 16-31 carry K={8..15,24..31} (ISA 7.12.2).
__device__ __forceinline__ h16v load_afrag(const _Float16* hbase, int lane, int kofs) {
  int m  = lane & 15;
  int ko = (lane & 16) ? 8 : 0;
  const _Float16* p = hbase + m * 64 + kofs + ko;
  H16 r;
  r.h[0] = *(const h8v*)p;
  r.h[1] = *(const h8v*)(p + 16);
  return r.v;
}

// B-fragment (32x16 f16, KxN): lane holds column n=lane&15; lanes 0-15 carry
// K={0..7,16..23}, lanes 16-31 carry K={8..15,24..31} (matches A-fragment).
__device__ __forceinline__ h16v load_bfrag(const _Float16* w, int lane, int tile,
                                           int ldk, int kofs) {
  int n  = lane & 15;
  int ko = (lane & 16) ? 8 : 0;
  const _Float16* p = w + (size_t)(tile * 16 + n) * ldk + kofs + ko;
  H16 r;
  r.h[0] = *(const h8v*)p;
  r.h[1] = *(const h8v*)(p + 16);
  return r.v;
}

extern "C" __global__ __launch_bounds__(BLK, 1)
void lstm2_coop(const float* __restrict__ x,
                const float* __restrict__ Wih1, const float* __restrict__ Whh1,
                const float* __restrict__ bih1, const float* __restrict__ bhh1,
                const float* __restrict__ Wih2, const float* __restrict__ Whh2,
                const float* __restrict__ bih2, const float* __restrict__ bhh2,
                const float* __restrict__ Wlin, const float* __restrict__ blin,
                float* __restrict__ out, int B, int T) {
  extern __shared__ char smemraw[];
  _Float16* W1    = (_Float16*)smemraw;          // [256][64]  W1[n][k] (B-major)
  _Float16* W2    = W1 + 256 * 64;               // [256][128] fused [W_ih2 W_hh2]
  float*    sb1   = (float*)(W2 + 256 * 128);    // [256] bias1 (padded)
  float*    sb2   = sb1 + 256;                   // [256] bias2 (padded)
  float*    swi   = sb2 + 256;                   // [256] W_ih1 column (padded)
  float*    swl   = swi + 256;                   // [64]  W_lin (padded)
  float*    ypart = swl + 64;                    // [2][64] per-wave y partials
  _Float16* Hb    = (_Float16*)(ypart + 128);    // h1[2], h2[2]: 4 x [16*64]

  const int tid = threadIdx.x;

  // ---- cooperative staging: pad gates 51->64 (N=256), K 51->64 / 102->128 ----
  for (int idx = tid; idx < 256 * 64; idx += BLK) {
    int n = idx >> 6, k = idx & 63;
    int q = n >> 6, j = n & 63;
    float w = (j < HH && k < HH) ? Whh1[(q * HH + j) * HH + k] : 0.f;
    W1[idx] = (_Float16)w;
  }
  for (int idx = tid; idx < 256 * 128; idx += BLK) {
    int n = idx >> 7, k = idx & 127;
    int q = n >> 6, j = n & 63;
    float w = 0.f;
    if (j < HH) {
      if (k < 64) { if (k < HH) w = Wih2[(q * HH + j) * HH + k]; }
      else        { int kk = k - 64; if (kk < HH) w = Whh2[(q * HH + j) * HH + kk]; }
    }
    W2[idx] = (_Float16)w;
  }
  for (int n = tid; n < 256; n += BLK) {
    int q = n >> 6, j = n & 63;
    int o = q * HH + j;
    bool v = (j < HH);
    sb1[n] = v ? (bih1[o] + bhh1[o]) : 0.f;
    sb2[n] = v ? (bih2[o] + bhh2[o]) : 0.f;
    swi[n] = v ? Wih1[o] : 0.f;
  }
  for (int k = tid; k < 64; k += BLK) swl[k] = (k < HH) ? Wlin[k] : 0.f;
  for (int idx = tid; idx < 4 * 16 * 64; idx += BLK) Hb[idx] = (_Float16)0.f;
  __syncthreads();

  const int wave = tid >> 5;   // wave g owns gate-column group g
  const int lane = tid & 31;
  const int hi   = lane >> 4;  // C-frag half: rows 0-7 vs 8-15
  const int nl   = lane & 15;  // C-frag column within a tile
  const int g    = wave;
  const int bbase = blockIdx.x * 16;

  // double buffers as integer offsets from Hb (keeps loads in LDS addrspace)
  // h1 buffers at Hb + {0, 1024}; h2 buffers at Hb + {2048, 3072}

  // per-lane constants for this wave's 4 tiles {g, g+4, g+8, g+12}
  float wi1r[4], b1r[4], b2r[4];
#pragma unroll
  for (int q = 0; q < 4; ++q) {
    int tt = 4 * q + g;
    wi1r[q] = swi[tt * 16 + nl];
    b1r[q]  = sb1[tt * 16 + nl];
    b2r[q]  = sb2[tt * 16 + nl];
  }
  const float wlg = swl[g * 16 + nl];  // W_lin weight for this lane's column
  const float bl  = blin[0];

  // persistent cell state, C-fragment layout: c[v] = c[row v+8*hi, col 16g+nl]
  float c1s[8], c2s[8];
#pragma unroll
  for (int v = 0; v < 8; ++v) { c1s[v] = 0.f; c2s[v] = 0.f; }

  // per-lane 32-bit element offsets into x for the 8 rows this lane tracks
  int xoff[8];
#pragma unroll
  for (int v = 0; v < 8; ++v) xoff[v] = (bbase + v + 8 * hi) * T;

  // prefetch x for t=0
  float xn[8];
#pragma unroll
  for (int v = 0; v < 8; ++v) xn[v] = x[xoff[v]];

#pragma unroll 1
  for (int t = 0; t < T; ++t) {
    const int rb = t & 1, wb = rb ^ 1;
    const int h1r = rb * 1024, h1w = wb * 1024;          // h1 buffer offsets
    const int h2r = 2048 + rb * 1024, h2w = 2048 + wb * 1024;

    float xv[8];
#pragma unroll
    for (int v = 0; v < 8; ++v) xv[v] = xn[v];
    int tnx = (t + 1 < T) ? (t + 1) : t;
#pragma unroll
    for (int v = 0; v < 8; ++v) xn[v] = x[xoff[v] + tnx];

    // ========== layer 1: gates = x*Wih1^T + h1*Whh1^T + b (this group) ==========
    h16v a0 = load_afrag(Hb + h1r, lane, 0);
    h16v a1 = load_afrag(Hb + h1r, lane, 32);
    {
      F8 acc[4];  // q = 0:i 1:f 2:g 3:o
#pragma unroll
      for (int q = 0; q < 4; ++q)
#pragma unroll
        for (int v = 0; v < 8; ++v) acc[q].f[v] = xv[v] * wi1r[q] + b1r[q];
#pragma unroll
      for (int q = 0; q < 4; ++q) {
        int tt = 4 * q + g;
        acc[q].v = __builtin_amdgcn_wmma_f32_16x16x32_f16(false, a0, false,
                     load_bfrag(W1, lane, tt, 64, 0),  (short)0, acc[q].v, false, false);
        acc[q].v = __builtin_amdgcn_wmma_f32_16x16x32_f16(false, a1, false,
                     load_bfrag(W1, lane, tt, 64, 32), (short)0, acc[q].v, false, false);
      }
#pragma unroll
      for (int v = 0; v < 8; ++v) {
        float cn = fsig(acc[1].f[v]) * c1s[v] + fsig(acc[0].f[v]) * ftanh(acc[2].f[v]);
        c1s[v] = cn;
        float hn = fsig(acc[3].f[v]) * ftanh(cn);
        Hb[h1w + (v + 8 * hi) * 64 + g * 16 + nl] = (_Float16)hn;
      }
    }

    __syncthreads();  // h1[wb] complete; h2[rb] stable; ypart[wb] complete

    // wave 0 drains y for step t-1 (off the other waves' critical path)
    if (wave == 0 && t > 0) {
      if (lane < 16) {
        const float* yp = ypart + wb * 64;  // wb == (t-1)&1
        float y = yp[nl] + yp[16 + nl] + yp[32 + nl] + yp[48 + nl] + bl;
        out[(size_t)(bbase + nl) * T + (t - 1)] = y;
      }
    }

    // ====== layer 2: gates = [h1 h2]*[Wih2 Whh2]^T + b (K=128, this group) ======
    h16v p0 = load_afrag(Hb + h1w, lane, 0);
    h16v p1 = load_afrag(Hb + h1w, lane, 32);
    h16v p2 = load_afrag(Hb + h2r, lane, 0);
    h16v p3 = load_afrag(Hb + h2r, lane, 32);
    {
      F8 acc[4];
#pragma unroll
      for (int q = 0; q < 4; ++q)
#pragma unroll
        for (int v = 0; v < 8; ++v) acc[q].f[v] = b2r[q];
#pragma unroll
      for (int q = 0; q < 4; ++q) {
        int tt = 4 * q + g;
        acc[q].v = __builtin_amdgcn_wmma_f32_16x16x32_f16(false, p0, false,
                     load_bfrag(W2, lane, tt, 128, 0),  (short)0, acc[q].v, false, false);
        acc[q].v = __builtin_amdgcn_wmma_f32_16x16x32_f16(false, p1, false,
                     load_bfrag(W2, lane, tt, 128, 32), (short)0, acc[q].v, false, false);
        acc[q].v = __builtin_amdgcn_wmma_f32_16x16x32_f16(false, p2, false,
                     load_bfrag(W2, lane, tt, 128, 64), (short)0, acc[q].v, false, false);
        acc[q].v = __builtin_amdgcn_wmma_f32_16x16x32_f16(false, p3, false,
                     load_bfrag(W2, lane, tt, 128, 96), (short)0, acc[q].v, false, false);
      }
      float pr[8];
#pragma unroll
      for (int v = 0; v < 8; ++v) {
        float cn = fsig(acc[1].f[v]) * c2s[v] + fsig(acc[0].f[v]) * ftanh(acc[2].f[v]);
        c2s[v] = cn;
        float hn = fsig(acc[3].f[v]) * ftanh(cn);
        Hb[h2w + (v + 8 * hi) * 64 + g * 16 + nl] = (_Float16)hn;
        pr[v] = hn * wlg;  // y partial for this lane's column
      }
      // reduce over the 16 columns of this group (stays within 16-lane halves)
#pragma unroll
      for (int v = 0; v < 8; ++v) {
        pr[v] += __shfl_xor(pr[v], 1, 32);
        pr[v] += __shfl_xor(pr[v], 2, 32);
        pr[v] += __shfl_xor(pr[v], 4, 32);
        pr[v] += __shfl_xor(pr[v], 8, 32);
      }
      if (nl == 0) {
        float* yp = ypart + rb * 64 + g * 16 + 8 * hi;
#pragma unroll
        for (int v = 0; v < 8; ++v) yp[v] = pr[v];
      }
    }
  }

  // drain y for the final step
  __syncthreads();
  if (wave == 0 && (tid & 31) < 16) {
    int nl0 = tid & 15;
    const float* yp = ypart + ((T - 1) & 1) * 64;
    float y = yp[nl0] + yp[16 + nl0] + yp[32 + nl0] + yp[48 + nl0] + bl;
    out[(size_t)(bbase + nl0) * T + (T - 1)] = y;
  }
}

extern "C" void kernel_launch(void* const* d_in, const int* in_sizes, int n_in,
                              void* d_out, int out_size, void* d_ws, size_t ws_size,
                              hipStream_t stream) {
  (void)n_in; (void)d_ws; (void)ws_size; (void)out_size;
  const float* x    = (const float*)d_in[0];
  const float* Wih1 = (const float*)d_in[1];
  const float* Whh1 = (const float*)d_in[2];
  const float* bih1 = (const float*)d_in[3];
  const float* bhh1 = (const float*)d_in[4];
  const float* Wih2 = (const float*)d_in[5];
  const float* Whh2 = (const float*)d_in[6];
  const float* bih2 = (const float*)d_in[7];
  const float* bhh2 = (const float*)d_in[8];
  const float* Wlin = (const float*)d_in[9];
  const float* blin = (const float*)d_in[10];
  float* out = (float*)d_out;

  const int B = 2048;
  const int T = in_sizes[0] / B;

  const size_t smem = (size_t)256 * 64 * 2        // W1 f16
                    + (size_t)256 * 128 * 2       // W2 f16 (fused)
                    + (256 * 3 + 64 + 128) * 4    // biases, wih1, wlin, ypart[2][64]
                    + (size_t)4 * 16 * 64 * 2;    // h1[2], h2[2] tiles
  (void)hipFuncSetAttribute((const void*)lstm2_coop,
                            hipFuncAttributeMaxDynamicSharedMemorySize, (int)smem);

  dim3 grid(B / 16);   // 128 blocks, one 16-row batch tile each
  dim3 block(BLK);     // 4 waves, one gate-column group each
  lstm2_coop<<<grid, block, smem, stream>>>(x, Wih1, Whh1, bih1, bhh1,
                                            Wih2, Whh2, bih2, bhh2,
                                            Wlin, blin, out, B, T);
}
